// RPN_63110249447707
// MI455X (gfx1250) — matrix operand order repaired and checked by reading, not compiled, not run on hardware
//
#include <hip/hip_runtime.h>
#include <math.h>

typedef __attribute__((ext_vector_type(8)))  __bf16 v8bf;
typedef __attribute__((ext_vector_type(16))) __bf16 v16bf;
typedef __attribute__((ext_vector_type(8)))  float  v8f;
typedef int v4i_ __attribute__((vector_size(16)));

#define NB    4
#define CIN   512
#define CMID  512
#define MSP   4096      /* 64*64 spatial positions */
#define NANCH 36864     /* MSP*9 */
#define NPAD  65536
#define NIN   6000
#define NOUT  300
#define NWRD  188       /* ceil(6000/32) */

/* d_out float offsets (scores, locs, rois, roi_idx, anchors) */
#define OFF_SCORES 0
#define OFF_LOCS   294912
#define OFF_ROIS   884736
#define OFF_IDX    889536
#define OFF_ANCH   890736

/* workspace byte offsets */
#define WS_XB   0u          /* bf16 x          16777216 */
#define WS_WP   16777216u   /* bf16 w packed    4718592 */
#define WS_WSL  21495808u   /* bf16 1x1 weights   65536 */
#define WS_H    21561344u   /* bf16 h          16777216 */
#define WS_RALL 38338560u   /* f32 rois_all     2359296 */
#define WS_KEYS 40697856u   /* u64 keys         2097152 */
#define WS_RS   42795008u   /* f32 rois sorted   384000 */
#define WS_MASK 43179008u   /* u32 nms mask    18048000 */

#if defined(__HIP_DEVICE_COMPILE__) && __has_builtin(__builtin_amdgcn_global_load_async_to_lds_b128)
#define USE_ASYNC 1
#else
#define USE_ASYNC 0
#endif

__device__ inline unsigned short f2bf(float f) {
  unsigned u = __float_as_uint(f);
  return (unsigned short)((u + 0x7FFFu + ((u >> 16) & 1u)) >> 16);
}

__device__ inline v16bf mkfrag(const unsigned short* lo, const unsigned short* hi) {
  v8bf a = *(const v8bf*)lo;
  v8bf b = *(const v8bf*)hi;
  v16bf r;
#pragma unroll
  for (int e = 0; e < 8; ++e) { r[e] = a[e]; r[8 + e] = b[e]; }
  return r;
}

#if USE_ASYNC
/* 16B global -> LDS async copy (no VGPR staging), tracked by ASYNCcnt */
__device__ inline void async_ld16(const unsigned short* g, unsigned short* l) {
  __builtin_amdgcn_global_load_async_to_lds_b128(
      (__attribute__((address_space(1))) v4i_*)g,
      (__attribute__((address_space(3))) v4i_*)l, 0, 0);
}
__device__ inline void wait_async0() {
#if __has_builtin(__builtin_amdgcn_s_wait_asynccnt)
  __builtin_amdgcn_s_wait_asynccnt(0);
#else
  asm volatile("s_wait_asynccnt 0" ::: "memory");
#endif
}
#endif

/* ---------------- packing / conversion ---------------- */

__global__ __launch_bounds__(256) void k_cvt_x(const float* __restrict__ x,
                                               unsigned short* __restrict__ xb, int n) {
  int g = blockIdx.x * 256 + threadIdx.x;
  if (g < n) xb[g] = f2bf(x[g]);
}

/* conv_w (O=512,C=512,3,3) -> wp[t][o][c] bf16 (K contiguous for async b128) */
__global__ __launch_bounds__(256) void k_pack_w(const float* __restrict__ w,
                                                unsigned short* __restrict__ wp) {
  int g = blockIdx.x * 256 + threadIdx.x;       /* < 9*512*512 */
  int t = g / (CIN * CMID);
  int r = g % (CIN * CMID);
  int o = r >> 9;
  int c = r & 511;
  wp[g] = f2bf(w[((size_t)o * CIN + c) * 9 + t]);
}

/* loc_w(36,512) + score_w(18,512) -> wsl[n][c] bf16 (rows 0..35 loc, 36..53 score) */
__global__ __launch_bounds__(256) void k_pack_wsl(const float* __restrict__ loc_w,
                                                  const float* __restrict__ score_w,
                                                  unsigned short* __restrict__ wsl) {
  int g = blockIdx.x * 256 + threadIdx.x;       /* < 64*512 */
  int n = g >> 9, c = g & 511;
  float v = 0.f;
  if (n < 36) v = loc_w[(size_t)n * CMID + c];
  else if (n < 54) v = score_w[(size_t)(n - 36) * CMID + c];
  wsl[g] = f2bf(v);
}

/* ---------------- 3x3 conv as implicit GEMM (WMMA bf16) ---------------- */
/* block tile 128M x 128N, 8 waves (4x2), wave tile 32M x 64N = 2x4 frags  */

__global__ __launch_bounds__(256)
void k_conv3x3(const unsigned short* __restrict__ xb, const unsigned short* __restrict__ wp,
               const float* __restrict__ convb, unsigned short* __restrict__ hb) {
  const int mt = blockIdx.x, nt = blockIdx.y, b = blockIdx.z;
  const int tid = threadIdx.x, lane = tid & 31, wv = tid >> 5;
  const int wm = wv >> 1, wn = wv & 1;
  __shared__ __align__(16) unsigned short lA[128 * 40];
  __shared__ __align__(16) unsigned short lB[128 * 40];

  v8f acc[2][4];
#pragma unroll
  for (int i = 0; i < 2; ++i)
#pragma unroll
    for (int j = 0; j < 4; ++j) acc[i][j] = (v8f){0.f,0.f,0.f,0.f,0.f,0.f,0.f,0.f};

  /* per-thread invariants for the A (im2col) loader:
     f = it*256+tid over a 32k x 128m tile  =>  ml = tid&127 (fixed),
     k = 2*it + (tid>>7); spatial position & OOB flag depend only on the tap. */
  const int ml = tid & 127;
  const int kh = tid >> 7;                /* 0 or 1 */
  const int msp = mt * 128 + ml;
  unsigned short* const dstA = &lA[ml * 40 + kh];

  for (int t = 0; t < 9; ++t) {
    const int dy = t / 3 - 1, dx = t % 3 - 1;
    const int y = (msp >> 6) + dy;
    const int xq = (msp & 63) + dx;
    const bool oob = ((unsigned)y >= 64u) || ((unsigned)xq >= 64u);
    /* wrapped (always in-bounds) spatial offset; OOB lanes select 0 below */
    const int sp = (((y & 63) << 6) | (xq & 63));
    for (int c0 = 0; c0 < CIN; c0 += 32) {
      /* A tile: branch-free — unconditional clamped loads + select-zero */
      {
        const unsigned short* src =
            xb + ((size_t)(b * CIN + c0 + kh) << 12) + sp;
#pragma unroll
        for (int it = 0; it < 16; ++it) {
          unsigned short v = src[(size_t)it << 13];   /* +2 channels per step */
          v = oob ? (unsigned short)0 : v;
          dstA[2 * it] = v;
        }
      }
      /* prefetch next channel chunk of x (global_prefetch_b8) */
      if (c0 + 32 < CIN) {
        const unsigned short* pf =
            xb + ((size_t)(b * CIN + c0 + 32 + (tid >> 3)) << 12) + mt * 128 + ((tid & 7) << 4);
        __builtin_prefetch(pf, 0, 1);
      }
      /* B tile: 128n x 32k from packed weights [t][o][c] (K contiguous) */
#if USE_ASYNC
#pragma unroll
      for (int it = 0; it < 2; ++it) {
        int f = it * 256 + tid;
        int row = f >> 2, ch = f & 3;         /* row 0..127, 16B chunk 0..3 */
        async_ld16(wp + (((size_t)(t * CMID + nt * 128 + row) << 9) + c0 + ch * 8),
                   &lB[row * 40 + ch * 8]);
      }
      wait_async0();
#else
      for (int it = 0; it < 16; ++it) {
        int f = it * 256 + tid;
        int k = f & 31, nl = f >> 5;
        lB[nl * 40 + k] = wp[(((size_t)(t * CMID + nt * 128 + nl)) << 9) + c0 + k];
      }
#endif
      __syncthreads();

      const int kbA = (lane >> 4) * 8;
      const int kbB = (lane >> 4) * 8;
      v16bf af[2], bf[4];
#pragma unroll
      for (int fm = 0; fm < 2; ++fm) {
        int row = wm * 32 + fm * 16 + (lane & 15);
        af[fm] = mkfrag(&lA[row * 40 + kbA], &lA[row * 40 + kbA + 16]);
      }
#pragma unroll
      for (int fn = 0; fn < 4; ++fn) {
        int col = wn * 64 + fn * 16 + (lane & 15);
        bf[fn] = mkfrag(&lB[col * 40 + kbB], &lB[col * 40 + kbB + 16]);
      }
#pragma unroll
      for (int fm = 0; fm < 2; ++fm)
#pragma unroll
        for (int fn = 0; fn < 4; ++fn)
          acc[fm][fn] = __builtin_amdgcn_wmma_f32_16x16x32_bf16(
              false, af[fm], false, bf[fn], (short)0, acc[fm][fn], false, false);
      __syncthreads();
    }
  }
  /* epilogue: bias + ReLU, store h[b][m][n] bf16 */
#pragma unroll
  for (int fm = 0; fm < 2; ++fm)
#pragma unroll
    for (int fn = 0; fn < 4; ++fn)
#pragma unroll
      for (int i = 0; i < 8; ++i) {
        int m = mt * 128 + wm * 32 + fm * 16 + i + 8 * (lane >> 4);
        int n = nt * 128 + wn * 64 + fn * 16 + (lane & 15);
        float v = acc[fm][fn][i] + convb[n];
        v = fmaxf(v, 0.f);
        hb[(((size_t)b * MSP + m) << 9) + n] = f2bf(v);
      }
}

/* ---------------- fused 1x1 heads (WMMA bf16, N=64) ---------------- */
/* block tile 256M x 64N, 8 waves, wave tile 32M x 64N = 2x4 frags     */

__global__ __launch_bounds__(256)
void k_fc1x1(const unsigned short* __restrict__ hb, const unsigned short* __restrict__ wsl,
             const float* __restrict__ score_b, const float* __restrict__ loc_b,
             float* __restrict__ out) {
  const int mt = blockIdx.x, b = blockIdx.z;
  const int tid = threadIdx.x, lane = tid & 31, wv = tid >> 5;
  __shared__ __align__(16) unsigned short lA[256 * 40];
  __shared__ __align__(16) unsigned short lB[64 * 40];

  v8f acc[2][4];
#pragma unroll
  for (int i = 0; i < 2; ++i)
#pragma unroll
    for (int j = 0; j < 4; ++j) acc[i][j] = (v8f){0.f,0.f,0.f,0.f,0.f,0.f,0.f,0.f};

  for (int c0 = 0; c0 < CMID; c0 += 32) {
#if USE_ASYNC
    /* A: one 256-row tile; each thread async-copies its row's 64 bytes */
#pragma unroll
    for (int ch = 0; ch < 4; ++ch)
      async_ld16(hb + ((((size_t)b * MSP + mt * 256 + tid) << 9) + c0 + ch * 8),
                 &lA[tid * 40 + ch * 8]);
    /* B: 64 rows x 64 bytes; thread t -> row t>>2, chunk t&3 */
    async_ld16(wsl + (((size_t)(tid >> 2) << 9) + c0 + (tid & 3) * 8),
               &lB[(tid >> 2) * 40 + (tid & 3) * 8]);
    wait_async0();
#else
    for (int it = 0; it < 32; ++it) {           /* A: 256m x 32k */
      int f = it * 256 + tid;
      int k = f & 31, ml = f >> 5;
      lA[ml * 40 + k] = hb[(((size_t)b * MSP + mt * 256 + ml) << 9) + c0 + k];
    }
    for (int it = 0; it < 8; ++it) {            /* B: 64n x 32k */
      int f = it * 256 + tid;
      int k = f & 31, nl = f >> 5;
      lB[nl * 40 + k] = wsl[((size_t)nl << 9) + c0 + k];
    }
#endif
    __syncthreads();

    const int kbA = (lane >> 4) * 8;
    const int kbB = (lane >> 4) * 8;
    v16bf af[2], bf[4];
#pragma unroll
    for (int fm = 0; fm < 2; ++fm) {
      int row = wv * 32 + fm * 16 + (lane & 15);
      af[fm] = mkfrag(&lA[row * 40 + kbA], &lA[row * 40 + kbA + 16]);
    }
#pragma unroll
    for (int fn = 0; fn < 4; ++fn) {
      int col = fn * 16 + (lane & 15);
      bf[fn] = mkfrag(&lB[col * 40 + kbB], &lB[col * 40 + kbB + 16]);
    }
#pragma unroll
    for (int fm = 0; fm < 2; ++fm)
#pragma unroll
      for (int fn = 0; fn < 4; ++fn)
        acc[fm][fn] = __builtin_amdgcn_wmma_f32_16x16x32_bf16(
            false, af[fm], false, bf[fn], (short)0, acc[fm][fn], false, false);
    __syncthreads();
  }
  /* epilogue: split into rpn_locs (n<36) and rpn_scores (36..53), fp32 + bias */
#pragma unroll
  for (int fm = 0; fm < 2; ++fm)
#pragma unroll
    for (int fn = 0; fn < 4; ++fn)
#pragma unroll
      for (int i = 0; i < 8; ++i) {
        int m = mt * 256 + wv * 32 + fm * 16 + i + 8 * (lane >> 4);
        int n = fn * 16 + (lane & 15);
        float v = acc[fm][fn][i];
        if (n < 36)
          out[OFF_LOCS + (size_t)b * (NANCH * 4) + (size_t)m * 36 + n] = v + loc_b[n];
        else if (n < 54)
          out[OFF_SCORES + (size_t)b * (NANCH * 2) + (size_t)m * 18 + (n - 36)] = v + score_b[n - 36];
      }
}

/* ---------------- proposal prep: anchors, decode, clip, keys ---------------- */

__global__ __launch_bounds__(256)
void k_prep(float* __restrict__ out, float* __restrict__ rall,
            unsigned long long* __restrict__ keys) {
  int g = blockIdx.x * 256 + threadIdx.x;       /* < 4*65536 */
  int b = g >> 16;
  int i = g & 0xFFFF;
  unsigned long long key = ~0ull;
  if (i < NANCH) {
    int m = i / 9, a = i % 9;
    int yq = m >> 6, xq = m & 63;
    const float RS[3] = {0.5f, 1.f, 2.f};
    const float SS[3] = {8.f, 16.f, 32.f};
    int ri = a / 3, si = a % 3;
    float ah = 16.f * SS[si] * sqrtf(RS[ri]);
    float aw = 16.f * SS[si] * sqrtf(1.f / RS[ri]);
    float ay1 = 8.f - 0.5f * ah + yq * 16.f;
    float ax1 = 8.f - 0.5f * aw + xq * 16.f;
    float ay2 = 8.f + 0.5f * ah + yq * 16.f;
    float ax2 = 8.f + 0.5f * aw + xq * 16.f;
    if (b == 0) {
      out[OFF_ANCH + (size_t)i * 4 + 0] = ay1;
      out[OFF_ANCH + (size_t)i * 4 + 1] = ax1;
      out[OFF_ANCH + (size_t)i * 4 + 2] = ay2;
      out[OFF_ANCH + (size_t)i * 4 + 3] = ax2;
    }
    const float* lp = out + OFF_LOCS + (size_t)b * (NANCH * 4) + (size_t)m * 36 + a * 4;
    float dyv = lp[0], dxv = lp[1], dhv = lp[2], dwv = lp[3];
    float hh = ay2 - ay1, wwv = ax2 - ax1;
    float cy = ay1 + 0.5f * hh, cx = ax1 + 0.5f * wwv;
    float cty = dyv * hh + cy, ctx = dxv * wwv + cx;
    float th = __expf(dhv) * hh, tw = __expf(dwv) * wwv;
    float y1 = fminf(fmaxf(cty - 0.5f * th, 0.f), 1024.f);
    float y2 = fminf(fmaxf(cty + 0.5f * th, 0.f), 1024.f);
    float x1 = fminf(fmaxf(ctx - 0.5f * tw, 0.f), 1024.f);
    float x2 = fminf(fmaxf(ctx + 0.5f * tw, 0.f), 1024.f);
    float* rp = rall + ((size_t)b * NANCH + i) * 4;
    rp[0] = y1; rp[1] = x1; rp[2] = y2; rp[3] = x2;
    bool valid = (y2 - y1 >= 16.f) && (x2 - x1 >= 16.f);
    const float* sp = out + OFF_SCORES + (size_t)b * (NANCH * 2) + (size_t)m * 18 + a * 2;
    float fg = 1.f / (1.f + __expf(sp[0] - sp[1]));   /* softmax fg */
    float sm = valid ? fg : -__builtin_inff();
    unsigned u = __float_as_uint(sm);
    u = (u & 0x80000000u) ? ~u : (u | 0x80000000u);   /* order-preserving map */
    key = ((unsigned long long)(~u) << 32) | (unsigned)i; /* ascending == desc score, asc idx */
  }
  keys[((size_t)b << 16) + i] = key;
}

/* ---------------- bitonic sort step (per-batch segments of 65536) ---------------- */

__global__ __launch_bounds__(256)
void k_bitonic(unsigned long long* __restrict__ keys, int j, int k) {
  int g = blockIdx.x * 256 + threadIdx.x;       /* < 4*65536 */
  int b = g >> 16;
  int i = g & 0xFFFF;
  int ixj = i ^ j;
  if (ixj > i) {
    unsigned long long* kb = keys + ((size_t)b << 16);
    unsigned long long a = kb[i], c = kb[ixj];
    bool asc = ((i & k) == 0);
    if ((a > c) == asc) { kb[i] = c; kb[ixj] = a; }
  }
}

/* ---------------- gather top-6000 sorted rois ---------------- */

__global__ __launch_bounds__(256)
void k_gather(const unsigned long long* __restrict__ keys,
              const float* __restrict__ rall, float* __restrict__ rs) {
  int g = blockIdx.x * 256 + threadIdx.x;
  if (g >= NB * NIN) return;
  int b = g / NIN, r = g % NIN;
  unsigned idx = (unsigned)(keys[((size_t)b << 16) + r] & 0xFFFFFFFFu);
  if (idx >= NANCH) idx = 0;
  const float* sp = rall + ((size_t)b * NANCH + idx) * 4;
  float* dp = rs + (size_t)g * 4;
  dp[0] = sp[0]; dp[1] = sp[1]; dp[2] = sp[2]; dp[3] = sp[3];
}

/* ---------------- NMS: parallel suppression mask ---------------- */

__global__ __launch_bounds__(32)
void k_nms_mask(const float* __restrict__ rs, unsigned* __restrict__ mask) {
  const int cb = blockIdx.x, rb = blockIdx.y, b = blockIdx.z;
  const int t = threadIdx.x;
  __shared__ float cy1[32], cx1[32], cy2[32], cx2[32];
  int j = cb * 32 + t;
  float a0 = 0.f, a1 = 0.f, a2 = 0.f, a3 = 0.f;
  if (j < NIN) {
    const float* p = rs + ((size_t)b * NIN + j) * 4;
    a0 = p[0]; a1 = p[1]; a2 = p[2]; a3 = p[3];
  }
  cy1[t] = a0; cx1[t] = a1; cy2[t] = a2; cx2[t] = a3;
  __syncthreads();
  int i = rb * 32 + t;
  if (i >= NIN) return;
  const float* p = rs + ((size_t)b * NIN + i) * 4;
  float ry1 = p[0], rx1 = p[1], ry2 = p[2], rx2 = p[3];
  float ra = (ry2 - ry1) * (rx2 - rx1);
  unsigned word = 0;
#pragma unroll
  for (int c = 0; c < 32; ++c) {
    int jg = cb * 32 + c;
    float yy1 = fmaxf(ry1, cy1[c]);
    float xx1 = fmaxf(rx1, cx1[c]);
    float yy2 = fminf(ry2, cy2[c]);
    float xx2 = fminf(rx2, cx2[c]);
    float inter = fmaxf(yy2 - yy1, 0.f) * fmaxf(xx2 - xx1, 0.f);
    float ca = (cy2[c] - cy1[c]) * (cx2[c] - cx1[c]);
    float iou = inter / (ra + ca - inter + 1e-9f);
    if (jg > i && jg < NIN && iou > 0.7f) word |= (1u << c);
  }
  mask[((size_t)b * NIN + i) * NWRD + cb] = word;
}

/* ---------------- NMS: serial greedy reduce, emit top-300 + indices ---------------- */

__global__ __launch_bounds__(256)
void k_nms_reduce(const float* __restrict__ rs, const unsigned* __restrict__ mask,
                  float* __restrict__ out) {
  const int b = blockIdx.x;
  const int tid = threadIdx.x;
  __shared__ unsigned remv[NWRD];
  __shared__ int rank;
  for (int w = tid; w < NWRD; w += 256) remv[w] = 0u;
  if (tid == 0) rank = 0;
  for (int g = tid; g < NOUT * 4; g += 256) out[OFF_ROIS + b * NOUT * 4 + g] = 0.f;
  int* ip = (int*)(out + OFF_IDX);
  for (int g = tid; g < NOUT; g += 256) ip[b * NOUT + g] = b;
  __syncthreads();
  for (int i = 0; i < NIN; ++i) {
    bool sup = (remv[i >> 5] >> (i & 31)) & 1u;
    const float* rp = rs + ((size_t)b * NIN + i) * 4;
    float y1 = rp[0], x1 = rp[1], y2 = rp[2], x2 = rp[3];
    bool keep = !sup && (y2 - y1 >= 16.f) && (x2 - x1 >= 16.f);
    int r = rank;
    __syncthreads();
    if (keep) {
      if (r < NOUT && tid < 4) out[OFF_ROIS + (b * NOUT + r) * 4 + tid] = rp[tid];
      const unsigned* mr = mask + ((size_t)b * NIN + i) * NWRD;
      for (int w = tid; w < NWRD; w += 256) remv[w] |= mr[w];
      if (tid == 0) rank = r + 1;
    }
    __syncthreads();
  }
}

/* ---------------- host ---------------- */

extern "C" void kernel_launch(void* const* d_in, const int* in_sizes, int n_in,
                              void* d_out, int out_size, void* d_ws, size_t ws_size,
                              hipStream_t stream) {
  (void)in_sizes; (void)n_in; (void)out_size; (void)ws_size;
  const float* x       = (const float*)d_in[0];
  const float* conv_w  = (const float*)d_in[1];
  const float* conv_b  = (const float*)d_in[2];
  const float* score_w = (const float*)d_in[3];
  const float* score_b = (const float*)d_in[4];
  const float* loc_w   = (const float*)d_in[5];
  const float* loc_b   = (const float*)d_in[6];
  float* out = (float*)d_out;
  char* ws = (char*)d_ws;
  unsigned short* xb  = (unsigned short*)(ws + WS_XB);
  unsigned short* wp  = (unsigned short*)(ws + WS_WP);
  unsigned short* wsl = (unsigned short*)(ws + WS_WSL);
  unsigned short* hb  = (unsigned short*)(ws + WS_H);
  float* rall = (float*)(ws + WS_RALL);
  unsigned long long* keys = (unsigned long long*)(ws + WS_KEYS);
  float* rs = (float*)(ws + WS_RS);
  unsigned* mask = (unsigned*)(ws + WS_MASK);

  k_cvt_x<<<(NB * CIN * MSP + 255) / 256, 256, 0, stream>>>(x, xb, NB * CIN * MSP);
  k_pack_w<<<(9 * CIN * CMID) / 256, 256, 0, stream>>>(conv_w, wp);
  k_pack_wsl<<<(CMID * 64) / 256, 256, 0, stream>>>(loc_w, score_w, wsl);
  k_conv3x3<<<dim3(32, 4, NB), 256, 0, stream>>>(xb, wp, conv_b, hb);
  k_fc1x1<<<dim3(16, 1, NB), 256, 0, stream>>>(hb, wsl, score_b, loc_b, out);
  k_prep<<<(NB * NPAD) / 256, 256, 0, stream>>>(out, rall, keys);
  for (int k = 2; k <= NPAD; k <<= 1)
    for (int j = k >> 1; j > 0; j >>= 1)
      k_bitonic<<<(NB * NPAD) / 256, 256, 0, stream>>>(keys, j, k);
  k_gather<<<(NB * NIN + 255) / 256, 256, 0, stream>>>(keys, rall, rs);
  k_nms_mask<<<dim3(NWRD, NWRD, NB), 32, 0, stream>>>(rs, mask);
  k_nms_reduce<<<NB, 256, 0, stream>>>(rs, mask, out);
}
